// DeltaModel_3204045603535
// MI455X (gfx1250) — matrix operand order repaired and checked by reading, not compiled, not run on hardware
//
#include <hip/hip_runtime.h>
#include <hip/hip_bf16.h>
#include <math.h>

// ---------------------------------------------------------------------------
// Model dims: B=256, L=2048, H=64, V=64, 2H=128
// ---------------------------------------------------------------------------
#define BB 256
#define LL 2048
#define HH 64
#define H2 128
#define NSTEP (LL - 1)              // 2047
#define NCHUNK ((NSTEP + 15) / 16)  // 128 (last chunk zero-padded)

typedef __attribute__((ext_vector_type(16))) _Float16 v16h;
typedef __attribute__((ext_vector_type(8)))  _Float16 v8h;
typedef __attribute__((ext_vector_type(8)))  float    v8f;
typedef __attribute__((ext_vector_type(4)))  int      i32x4;

// ---------------------------------------------------------------------------
// gfx1250 async global->LDS copy (ASYNCcnt path), with safe fallback.
// Probe-derived signature: (i32x4 addrspace(1)*, i32x4 addrspace(3)*, imm, imm)
// ---------------------------------------------------------------------------
#if __has_builtin(__builtin_amdgcn_global_load_async_to_lds_b128)
#define ASYNC_LDS 1
#endif

__device__ inline void async_ld16(const float* g, float* l) {
#ifdef ASYNC_LDS
  __builtin_amdgcn_global_load_async_to_lds_b128(
      (__attribute__((address_space(1))) i32x4*)(float*)g,
      (__attribute__((address_space(3))) i32x4*)l, 0, 0);
#else
  l[0] = g[0]; l[1] = g[1]; l[2] = g[2]; l[3] = g[3];
#endif
}

__device__ inline void wait_async0() {
#ifdef ASYNC_LDS
#if __has_builtin(__builtin_amdgcn_s_wait_asynccnt)
  __builtin_amdgcn_s_wait_asynccnt(0);
#else
  asm volatile("s_wait_asynccnt 0x0" ::: "memory");
#endif
#endif
}

// ---------------------------------------------------------------------------
// WMMA operand builders (wave32, V_WMMA_F32_16X16X32_F16 layouts per ISA 7.12.2)
//
// A (16x32 f16): lane<16 -> row=lane, halves = K {kb+0..7, kb+16..23}
//                lane>=16 -> row=lane-16, halves = K {kb+8..15, kb+24..31}
// B (32x16 f16): lane<16 -> col=lane,    halves = K {kb+0..7, kb+16..23}
//                lane>=16 -> col=lane-16, halves = K {kb+8..15, kb+24..31}
// C/D (16x16 f32): c[v] = D[ v + (lane>>4)*8 ][ lane&15 ]
// ---------------------------------------------------------------------------
__device__ inline v16h load_A(const _Float16* base, int ldk, int kbase, int lane) {
  int row = lane & 15;
  int kh  = (lane >> 4) << 3;                  // 0 or 8
  const v8h* p0 = (const v8h*)(base + row * ldk + kbase + kh);
  const v8h* p1 = (const v8h*)(base + row * ldk + kbase + kh + 16);
  v8h lo = *p0, hi = *p1;
  v16h a;
#pragma unroll
  for (int i = 0; i < 8; ++i) { a[i] = lo[i]; a[8 + i] = hi[i]; }
  return a;
}

__device__ inline v16h load_B(const _Float16* w, int ldk, int nbase, int kbase, int lane) {
  int n = nbase + (lane & 15);
  int k = kbase + ((lane >> 4) << 3);          // 0 or 8
  const v8h* p = (const v8h*)(w + n * ldk + k);
  v8h lo = p[0], hi = p[2];
  v16h b;
#pragma unroll
  for (int i = 0; i < 8; ++i) { b[i] = lo[i]; b[8 + i] = hi[i]; }
  return b;
}

#define WMMA_F16(a, b, c) \
  __builtin_amdgcn_wmma_f32_16x16x32_f16(false, (a), false, (b), (short)0, (c), false, false)

// ---------------------------------------------------------------------------
// Kernel: fp32 -> fp16 weight conversion (row-major copy == B-operand layout)
// ---------------------------------------------------------------------------
__global__ void cvt_kernel(const float* __restrict__ src, _Float16* __restrict__ dst, int n) {
  int i = blockIdx.x * 256 + threadIdx.x;
  if (i < n) dst[i] = (_Float16)src[i];
}

// ---------------------------------------------------------------------------
// Kernel: encode 16 tokens per wave: gather -> FFN(WMMA) -> LN -> k/v(WMMA)
// grid = B*L/16 blocks of 32 threads (one wave, EXEC all ones for WMMA)
// ---------------------------------------------------------------------------
__global__ void encode_kernel(const int* __restrict__ seq, const float* __restrict__ embed,
                              const float* __restrict__ b1, const float* __restrict__ b2,
                              const float* __restrict__ gamma, const float* __restrict__ beta,
                              const _Float16* __restrict__ W1h, const _Float16* __restrict__ W2h,
                              const _Float16* __restrict__ Wkh, const _Float16* __restrict__ Wvh,
                              float* __restrict__ kn_out, float* __restrict__ v_out,
                              float* __restrict__ hs_last) {
  __shared__ float    sE[16][HH];
  __shared__ _Float16 sEh[16][HH];
  __shared__ _Float16 sZh[16][H2];
  __shared__ float    sHS[16][HH];
  __shared__ _Float16 sHSh[16][HH];
  __shared__ float    sK[16][HH];
  __shared__ float    sV[16][HH];
  __shared__ float    sScale[16];

  const int lane = threadIdx.x;
  const int gr0  = blockIdx.x * 16;       // flat row base in [B*L]
  const int b    = gr0 >> 11;             // / L
  const int l0   = gr0 & (LL - 1);        // % L

  // ---- gather embeddings for 16 tokens ----
  for (int idx = lane; idx < 16 * HH; idx += 32) {
    int row = idx >> 6, col = idx & 63;
    int tok = seq[gr0 + row];
    float e = embed[tok * HH + col];
    sE[row][col]  = e;
    sEh[row][col] = (_Float16)e;
  }
  __syncthreads();

  // ---- GEMM1: Z = relu(E @ W1^T + b1)  [16 x 128] ----
  for (int nt = 0; nt < 8; ++nt) {
    v8f acc = {};
#pragma unroll
    for (int kt = 0; kt < 2; ++kt) {
      v16h a  = load_A(&sEh[0][0], HH, kt * 32, lane);
      v16h bm = load_B(W1h, HH, nt * 16, kt * 32, lane);
      acc = WMMA_F16(a, bm, acc);
    }
    int n = nt * 16 + (lane & 15);
    float bias = b1[n];
#pragma unroll
    for (int v = 0; v < 8; ++v) {
      int mrow = v + ((lane >> 4) << 3);
      sZh[mrow][n] = (_Float16)fmaxf(acc[v] + bias, 0.f);
    }
  }
  __syncthreads();

  // ---- GEMM2: H = E + Z @ W2^T + b2  [16 x 64] ----
  for (int nt = 0; nt < 4; ++nt) {
    v8f acc = {};
#pragma unroll
    for (int kt = 0; kt < 4; ++kt) {
      v16h a  = load_A(&sZh[0][0], H2, kt * 32, lane);
      v16h bm = load_B(W2h, H2, nt * 16, kt * 32, lane);
      acc = WMMA_F16(a, bm, acc);
    }
    int n = nt * 16 + (lane & 15);
    float bias = b2[n];
#pragma unroll
    for (int v = 0; v < 8; ++v) {
      int mrow = v + ((lane >> 4) << 3);
      sHS[mrow][n] = sE[mrow][n] + acc[v] + bias;   // pre-LN h
    }
  }
  __syncthreads();

  // ---- LayerNorm (biased var, eps=1e-5); lanes 0..15 own one row each ----
  if (lane < 16) {
    float mu = 0.f;
#pragma unroll
    for (int j = 0; j < HH; ++j) mu += sHS[lane][j];
    mu *= (1.f / HH);
    float var = 0.f;
#pragma unroll
    for (int j = 0; j < HH; ++j) { float d = sHS[lane][j] - mu; var += d * d; }
    var *= (1.f / HH);
    float inv = rsqrtf(var + 1e-5f);
#pragma unroll
    for (int j = 0; j < HH; ++j) {
      float hsv = (sHS[lane][j] - mu) * inv * gamma[j] + beta[j];
      sHS[lane][j]  = hsv;
      sHSh[lane][j] = (_Float16)hsv;
    }
  }
  __syncthreads();

  // ---- K/V projections (WMMA), f32 accumulate ----
  for (int nt = 0; nt < 4; ++nt) {
    v8f accK = {}, accV = {};
#pragma unroll
    for (int kt = 0; kt < 2; ++kt) {
      v16h a  = load_A(&sHSh[0][0], HH, kt * 32, lane);
      v16h bk = load_B(Wkh, HH, nt * 16, kt * 32, lane);
      v16h bv = load_B(Wvh, HH, nt * 16, kt * 32, lane);
      accK = WMMA_F16(a, bk, accK);
      accV = WMMA_F16(a, bv, accV);
    }
    int n = nt * 16 + (lane & 15);
#pragma unroll
    for (int v = 0; v < 8; ++v) {
      int mrow = v + ((lane >> 4) << 3);
      sK[mrow][n] = accK[v];
      sV[mrow][n] = accV[v];
    }
  }
  __syncthreads();

  // ---- F.normalize(k): 1 / max(||k||, 1e-12) per row ----
  if (lane < 16) {
    float ss = 0.f;
#pragma unroll
    for (int j = 0; j < HH; ++j) ss += sK[lane][j] * sK[lane][j];
    sScale[lane] = 1.f / fmaxf(sqrtf(ss), 1e-12f);
  }
  __syncthreads();

  // ---- stream out k_n, v in [t][b][h] layout (coalesced for the scan) ----
  for (int idx = lane; idx < 16 * HH; idx += 32) {
    int row = idx >> 6, col = idx & 63;
    int l = l0 + row;
    if (l < LL - 1) {
      size_t o = ((size_t)l * BB + b) * HH + col;
      kn_out[o] = sK[row][col] * sScale[row];
      v_out[o]  = sV[row][col];
    } else {
      hs_last[b * HH + col] = sHS[row][col];   // last token's hs for q
    }
  }
}

// ---------------------------------------------------------------------------
// Kernel: generic Y[M x 64] = X[M x 64] @ W^T(+bias) via WMMA, 16 rows/block
// ---------------------------------------------------------------------------
__global__ void gemm16_kernel(const float* __restrict__ X, const _Float16* __restrict__ Wh,
                              const float* __restrict__ bias, float* __restrict__ Y,
                              int hasBias) {
  __shared__ _Float16 sX[16][HH];
  const int lane = threadIdx.x;
  const int r0   = blockIdx.x * 16;
  for (int idx = lane; idx < 16 * HH; idx += 32)
    sX[idx >> 6][idx & 63] = (_Float16)X[(r0 + (idx >> 6)) * HH + (idx & 63)];
  __syncthreads();
  for (int nt = 0; nt < 4; ++nt) {
    v8f acc = {};
#pragma unroll
    for (int kt = 0; kt < 2; ++kt) {
      v16h a = load_A(&sX[0][0], HH, kt * 32, lane);
      v16h b = load_B(Wh, HH, nt * 16, kt * 32, lane);
      acc = WMMA_F16(a, b, acc);
    }
    int n = nt * 16 + (lane & 15);
    float bb = hasBias ? bias[n] : 0.f;
#pragma unroll
    for (int v = 0; v < 8; ++v) {
      int m = v + ((lane >> 4) << 3);
      Y[(r0 + m) * HH + n] = acc[v] + bb;
    }
  }
}

// ---------------------------------------------------------------------------
// Kernel: CHUNKED delta-rule scan (WMMA). One block (128 threads = 4 waves)
// per batch element. State M (64x64) lives in LDS as f32 + split-f16 (hi/lo).
//
// Per 16-step chunk:
//   T  = K @ M^T            (WMMA, hi+lo for ~f32 accuracy)
//   Bm = V - T ; S = K K^T  (WMMA)
//   (I + strict_lower(S)) U = Bm   (forward substitution, 64 indep. columns)
//   M += U^T K              (WMMA, K-dim padded 16->32, f32 C accumulate)
// Serial chain: 128 chunks instead of 2047 steps.
// V chunks are DMA'd global->LDS asynchronously into a double buffer
// (ASYNCcnt); K rides the register path since it needs f32->f16 + transpose.
// ---------------------------------------------------------------------------
__global__ void scan_kernel(const float* __restrict__ kn, const float* __restrict__ vv,
                            const float* __restrict__ q, float* __restrict__ Mq) {
  __shared__ _Float16 sKh[16][HH];     // K rows (A-operand layout)
  __shared__ _Float16 sKt[HH][32];     // K^T, K-dim padded to 32 (B-operand layout)
  __shared__ float    sV [2][16][HH];  // V rows, double-buffered (async DMA dest)
  __shared__ float    sT [16][HH];     // T, then Bm = V - T in place
  __shared__ float    sS [16][16];     // S = K K^T
  __shared__ _Float16 sUt[HH][32];     // U^T, K-dim padded to 32 (A-operand layout)
  __shared__ float    sM  [HH][HH];    // state, f32
  __shared__ _Float16 sMhi[HH][HH];    // state, f16 high part
  __shared__ _Float16 sMlo[HH][HH];    // state, f16 residual

  const int b    = blockIdx.x;
  const int tid  = threadIdx.x;        // 0..127
  const int w    = tid >> 5;           // wave id 0..3
  const int lane = tid & 31;

  // ---- init state and pad regions ----
  for (int i = tid; i < HH * HH; i += 128) {
    sM[0][i]   = 0.f;
    sMhi[0][i] = (_Float16)0.f;
    sMlo[0][i] = (_Float16)0.f;
  }
  if (tid < HH) {
#pragma unroll
    for (int j = 16; j < 32; ++j) { sKt[tid][j] = (_Float16)0.f; sUt[tid][j] = (_Float16)0.f; }
  }

  // ---- issue async V DMA for chunk 0; load K chunk 0 into registers ----
  {
#pragma unroll
    for (int j = 0; j < 2; ++j) {
      int cc = tid + j * 128;                 // 16-byte chunk id, 0..255
      int i = cc >> 4, h = (cc & 15) * 4;     // row, col
      if (i < NSTEP) async_ld16(&vv[((size_t)i * BB + b) * HH + h], &sV[0][i][h]);
      else { float* l = &sV[0][i][h]; l[0] = l[1] = l[2] = l[3] = 0.f; }
    }
  }
  float kreg[8];
#pragma unroll
  for (int j = 0; j < 8; ++j) {
    int f = tid * 8 + j, i = f >> 6, h = f & 63;
    kreg[j] = (i < NSTEP) ? kn[((size_t)i * BB + b) * HH + h] : 0.f;
  }

  for (int c = 0; c < NCHUNK; ++c) {
    const int buf = c & 1;
    // ---- ensure async V for this chunk has landed; stage K (both layouts) --
    wait_async0();
#pragma unroll
    for (int j = 0; j < 8; ++j) {
      int f = tid * 8 + j, i = f >> 6, h = f & 63;
      _Float16 kh = (_Float16)kreg[j];
      sKh[i][h] = kh;
      sKt[h][i] = kh;
    }
    __syncthreads();

    // ---- issue async V DMA + register K prefetch for chunk c+1 ----
    if (c + 1 < NCHUNK) {
#pragma unroll
      for (int j = 0; j < 2; ++j) {
        int cc = tid + j * 128;
        int i = cc >> 4, h = (cc & 15) * 4;
        int t = (c + 1) * 16 + i;
        if (t < NSTEP) async_ld16(&vv[((size_t)t * BB + b) * HH + h], &sV[buf ^ 1][i][h]);
        else { float* l = &sV[buf ^ 1][i][h]; l[0] = l[1] = l[2] = l[3] = 0.f; }
      }
#pragma unroll
      for (int j = 0; j < 8; ++j) {
        int f = tid * 8 + j, i = f >> 6, h = f & 63;
        int t = (c + 1) * 16 + i;
        kreg[j] = (t < NSTEP) ? kn[((size_t)t * BB + b) * HH + h] : 0.f;
      }
    }

    // ---- T = K @ M^T : wave w computes column tile w (hi + lo parts) ----
    {
      v8f acc = {};
#pragma unroll
      for (int kt = 0; kt < 2; ++kt) {
        v16h a   = load_A(&sKh[0][0], HH, kt * 32, lane);
        v16h bhi = load_B(&sMhi[0][0], HH, w * 16, kt * 32, lane);
        v16h blo = load_B(&sMlo[0][0], HH, w * 16, kt * 32, lane);
        acc = WMMA_F16(a, bhi, acc);
        acc = WMMA_F16(a, blo, acc);
      }
      int n = w * 16 + (lane & 15);
#pragma unroll
      for (int v = 0; v < 8; ++v) sT[v + ((lane >> 4) << 3)][n] = acc[v];
    }
    // ---- S = K K^T : wave 0 only ----
    if (w == 0) {
      v8f acc = {};
#pragma unroll
      for (int kt = 0; kt < 2; ++kt) {
        v16h a  = load_A(&sKh[0][0], HH, kt * 32, lane);
        v16h bb = load_B(&sKh[0][0], HH, 0, kt * 32, lane);
        acc = WMMA_F16(a, bb, acc);
      }
      int n = lane & 15;
#pragma unroll
      for (int v = 0; v < 8; ++v) sS[v + ((lane >> 4) << 3)][n] = acc[v];
    }
    __syncthreads();

    // ---- Bm = V - T (in place into sT) ----
#pragma unroll
    for (int j = 0; j < 8; ++j) {
      int f = tid * 8 + j, i = f >> 6, h = f & 63;
      sT[i][h] = sV[buf][i][h] - sT[i][h];
    }
    __syncthreads();

    // ---- forward substitution: column h independent, threads 0..63 ----
    if (tid < HH) {
      const int h = tid;
      float u[16];
#pragma unroll
      for (int i = 0; i < 16; ++i) {
        float x = sT[i][h];
#pragma unroll
        for (int j = 0; j < 16; ++j) {
          if (j < i) x -= sS[i][j] * u[j];
        }
        u[i] = x;
        sUt[h][i] = (_Float16)x;
      }
    }
    __syncthreads();

    // ---- M += U^T K : wave w owns row tile w; 4 column tiles each ----
    for (int nt = 0; nt < 4; ++nt) {
      int n = nt * 16 + (lane & 15);
      v8f cc;
#pragma unroll
      for (int v = 0; v < 8; ++v) cc[v] = sM[w * 16 + v + ((lane >> 4) << 3)][n];
      v16h a  = load_A(&sUt[w * 16][0], 32, 0, lane);
      v16h bb = load_B(&sKt[0][0], 32, nt * 16, 0, lane);
      cc = WMMA_F16(a, bb, cc);
#pragma unroll
      for (int v = 0; v < 8; ++v) {
        int mr = w * 16 + v + ((lane >> 4) << 3);
        float val = cc[v];
        _Float16 hi = (_Float16)val;
        sM[mr][n]   = val;
        sMhi[mr][n] = hi;
        sMlo[mr][n] = (_Float16)(val - (float)hi);
      }
    }
    __syncthreads();
  }

  // ---- Mq = M @ q (64 independent dot products) ----
  if (tid < HH) {
    float p = 0.f;
#pragma unroll
    for (int j = 0; j < HH; ++j) p += sM[tid][j] * q[b * HH + j];
    Mq[b * HH + tid] = p;
  }
}

// ---------------------------------------------------------------------------
// Host launcher
// ---------------------------------------------------------------------------
extern "C" void kernel_launch(void* const* d_in, const int* in_sizes, int n_in,
                              void* d_out, int out_size, void* d_ws, size_t ws_size,
                              hipStream_t stream) {
  const int*   seq   = (const int*)d_in[0];
  const float* embed = (const float*)d_in[1];
  const float* W1    = (const float*)d_in[2];
  const float* b1    = (const float*)d_in[3];
  const float* W2    = (const float*)d_in[4];
  const float* b2    = (const float*)d_in[5];
  const float* gamma = (const float*)d_in[6];
  const float* beta  = (const float*)d_in[7];
  const float* Wk    = (const float*)d_in[8];
  const float* Wv    = (const float*)d_in[9];
  const float* Wq    = (const float*)d_in[10];
  const float* Wrp   = (const float*)d_in[11];
  const float* brp   = (const float*)d_in[12];
  const float* Wout  = (const float*)d_in[13];
  const float* bout  = (const float*)d_in[14];

  // workspace layout
  float* ws = (float*)d_ws;
  const size_t KNV = (size_t)NSTEP * BB * HH;   // 33,538,048 floats
  float* kn  = ws;
  float* vv  = kn + KNV;
  float* hsl = vv + KNV;
  float* q   = hsl + BB * HH;
  float* Mq  = q + BB * HH;
  float* r   = Mq + BB * HH;
  _Float16* W1h   = (_Float16*)(r + BB * HH);
  _Float16* W2h   = W1h + H2 * HH;
  _Float16* Wkh   = W2h + HH * H2;
  _Float16* Wvh   = Wkh + HH * HH;
  _Float16* Wqh   = Wvh + HH * HH;
  _Float16* Wrph  = Wqh + HH * HH;
  _Float16* Wouth = Wrph + HH * HH;

  // weight conversions (f16 row-major == WMMA B-operand friendly layout)
  cvt_kernel<<<32, 256, 0, stream>>>(W1,   W1h,   H2 * HH);
  cvt_kernel<<<32, 256, 0, stream>>>(W2,   W2h,   HH * H2);
  cvt_kernel<<<16, 256, 0, stream>>>(Wk,   Wkh,   HH * HH);
  cvt_kernel<<<16, 256, 0, stream>>>(Wv,   Wvh,   HH * HH);
  cvt_kernel<<<16, 256, 0, stream>>>(Wq,   Wqh,   HH * HH);
  cvt_kernel<<<16, 256, 0, stream>>>(Wrp,  Wrph,  HH * HH);
  cvt_kernel<<<16, 256, 0, stream>>>(Wout, Wouth, HH * HH);

  // encode + k/v projections
  encode_kernel<<<(BB * LL) / 16, 32, 0, stream>>>(seq, embed, b1, b2, gamma, beta,
                                                   W1h, W2h, Wkh, Wvh, kn, vv, hsl);

  // q = hs_last @ Wq^T
  gemm16_kernel<<<BB / 16, 32, 0, stream>>>(hsl, Wqh, nullptr, q, 0);

  // chunked delta-rule scan (WMMA + async V DMA) + Mq = M q
  scan_kernel<<<BB, 128, 0, stream>>>(kn, vv, q, Mq);

  // r = Mq @ Wrp^T + brp ; out = r @ Wout^T + bout
  gemm16_kernel<<<BB / 16, 32, 0, stream>>>(Mq, Wrph, brp, r, 1);
  gemm16_kernel<<<BB / 16, 32, 0, stream>>>(r, Wouth, bout, (float*)d_out, 1);
}
